// AxialAttentionBlock_9818295238705
// MI455X (gfx1250) — matrix operand order repaired and checked by reading, not compiled, not run on hardware
//
#include <hip/hip_runtime.h>
#include <math.h>

// ---------------------------------------------------------------------------
// Types / helpers
// ---------------------------------------------------------------------------
typedef __attribute__((ext_vector_type(16))) __bf16       v16bf;
typedef __attribute__((ext_vector_type(8)))  float        v8f;
typedef __attribute__((ext_vector_type(4)))  unsigned int u32x4;
typedef __attribute__((ext_vector_type(4)))  int          v4i;

union FragU { u32x4 q[2]; v16bf v; };

__device__ __forceinline__ unsigned short f2bf(float f) {
  union { float f; unsigned int u; } x; x.f = f;
  unsigned int r = x.u + 0x7fffu + ((x.u >> 16) & 1u);   // RNE
  return (unsigned short)(r >> 16);
}
__device__ __forceinline__ float bf2f(unsigned short u) {
  union { unsigned int u; float f; } x; x.u = ((unsigned int)u) << 16;
  return x.f;
}
// Two 16-byte chunks -> one 16-element bf16 fragment register set
__device__ __forceinline__ v16bf ldfrag(const unsigned short* p0,
                                        const unsigned short* p1) {
  FragU f;
  f.q[0] = *(const u32x4*)p0;
  f.q[1] = *(const u32x4*)p1;
  return f.v;
}

// ---- CDNA5 async global->LDS copy (ASYNCcnt) ------------------------------
__device__ __forceinline__ void async_copy_b128(const unsigned short* g,
                                                unsigned short* l) {
#if __has_builtin(__builtin_amdgcn_global_load_async_to_lds_b128)
  __builtin_amdgcn_global_load_async_to_lds_b128(
      (__attribute__((address_space(1))) v4i*)(void*)g,
      (__attribute__((address_space(3))) v4i*)(void*)l, 0, 0);
#else
  // VDST = LDS byte address (low 32 bits of generic LDS addr), VADDR = 64-bit
  asm volatile("global_load_async_to_lds_b128 %0, %1, off"
               :: "v"((unsigned)(unsigned long long)l),
                  "v"((unsigned long long)g)
               : "memory");
#endif
}
__device__ __forceinline__ void wait_async_3() {
#if __has_builtin(__builtin_amdgcn_s_wait_asynccnt)
  __builtin_amdgcn_s_wait_asynccnt(3);
#else
  asm volatile("s_wait_asynccnt 0x3" ::: "memory");
#endif
}
__device__ __forceinline__ void wait_async_0() {
#if __has_builtin(__builtin_amdgcn_s_wait_asynccnt)
  __builtin_amdgcn_s_wait_asynccnt(0);
#else
  asm volatile("s_wait_asynccnt 0x0" ::: "memory");
#endif
}

// Problem constants
constexpr int Bn    = 2;
constexpr int Hn    = 128;
constexpr int Wn    = 128;
constexpr int HWn   = Hn * Wn;          // 16384
constexpr int Pn    = Bn * HWn;         // 32768 positions
constexpr int Cn    = 768;
constexpr int NHn   = 8;
constexpr int HEADn = 96;
constexpr int QKVCn = 3 * Cn;           // 2304
constexpr int FFn   = 4 * Cn;           // 3072
constexpr int Ln    = 128;              // attention sequence length (axial)
constexpr int BKn   = 32;               // GEMM K-stage (one wmma K step)

// ---------------------------------------------------------------------------
// f32 -> bf16 weight conversion
// ---------------------------------------------------------------------------
__launch_bounds__(256)
__global__ void k_cvt_bf16(const float* __restrict__ w,
                           unsigned short* __restrict__ o, int n) {
  int i = blockIdx.x * 256 + threadIdx.x;
  if (i < n) o[i] = f2bf(w[i]);
}

// ---------------------------------------------------------------------------
// Per-(b,c) unbiased std.  layout 0: bchw contiguous; layout 1: [pos, C]
// ---------------------------------------------------------------------------
__launch_bounds__(256)
__global__ void k_std(const float* __restrict__ src, float* __restrict__ stds,
                      int layout) {
  int bc = blockIdx.x;                  // b*C + c
  int b = bc / Cn, c = bc % Cn;
  size_t base; size_t stride;
  if (layout == 0) { base = (size_t)bc * HWn;            stride = 1;  }
  else             { base = (size_t)b * HWn * Cn + c;    stride = Cn; }
  float s = 0.f, s2 = 0.f;
  for (int i = threadIdx.x; i < HWn; i += 256) {
    float v = src[base + (size_t)i * stride];
    s += v; s2 += v * v;
  }
  __shared__ float sh[256], sh2[256];
  sh[threadIdx.x] = s; sh2[threadIdx.x] = s2;
  __syncthreads();
  for (int d = 128; d > 0; d >>= 1) {
    if (threadIdx.x < d) { sh[threadIdx.x] += sh[threadIdx.x + d];
                           sh2[threadIdx.x] += sh2[threadIdx.x + d]; }
    __syncthreads();
  }
  if (threadIdx.x == 0) {
    float sum = sh[0], sumsq = sh2[0];
    float mean = sum / (float)HWn;
    float var = (sumsq - sum * mean) / (float)(HWn - 1);  // ddof=1
    stds[bc] = sqrtf(fmaxf(var, 0.f));
  }
}

// ---------------------------------------------------------------------------
// RMS-instance-norm: x(bchw) -> bf16 [pos, C]
// ---------------------------------------------------------------------------
__launch_bounds__(256)
__global__ void k_norm_bchw(const float* __restrict__ x,
                            const float* __restrict__ stds,
                            const float* __restrict__ w,
                            unsigned short* __restrict__ out) {
  long long i = (long long)blockIdx.x * 256 + threadIdx.x;
  if (i >= (long long)Bn * Cn * HWn) return;
  int hw = (int)(i % HWn);
  long long t = i / HWn;
  int c = (int)(t % Cn);
  int b = (int)(t / Cn);
  float v = x[i] / (stds[b * Cn + c] + 1e-8f) * w[c];
  out[((size_t)b * HWn + hw) * Cn + c] = f2bf(v);
}

// RMS-instance-norm on [pos, C] f32 -> bf16 [pos, C] (same layout)
__launch_bounds__(256)
__global__ void k_norm_posC(const float* __restrict__ a,
                            const float* __restrict__ stds,
                            const float* __restrict__ w,
                            unsigned short* __restrict__ out) {
  long long i = (long long)blockIdx.x * 256 + threadIdx.x;
  if (i >= (long long)Pn * Cn) return;
  int c = (int)(i % Cn);
  long long pos = i / Cn;
  int b = (int)(pos >> 14);             // pos / HWn
  out[i] = f2bf(a[i] / (stds[b * Cn + c] + 1e-8f) * w[c]);
}

// ---------------------------------------------------------------------------
// WMMA GEMM, async-LDS double-buffered:
//   D[M,N] = A[M,K](bf16,row) * Bw[N,K](bf16,row)^T  (+ epilogue)
//   block tile 128x64, 8 waves (4x2), wave tile 32x32 (2x2 of 16x16)
//   K staged through LDS in BK=32 slices via GLOBAL_LOAD_ASYNC_TO_LDS_B128,
//   double buffered, ASYNCcnt-synchronized (3 copies/thread/stage).
//   mode 0: bf16 out = acc+bias                      (qkv)
//   mode 1: x2 = (acc+bias)*gamma + resid ; f32+bf16 (out proj + residual)
//   mode 2: bf16 out = gelu_exact(acc+bias)          (fc1)
//   mode 3: f32 out = acc+bias                       (fc2)
// ---------------------------------------------------------------------------
__launch_bounds__(256)
__global__ void k_gemm(const unsigned short* __restrict__ A,
                       const unsigned short* __restrict__ Bw,
                       const float* __restrict__ bias,
                       float* __restrict__ outF,
                       unsigned short* __restrict__ outBf,
                       const float* __restrict__ gamma,
                       const float* __restrict__ resid,
                       int M, int N, int K, int mode) {
  __shared__ unsigned short sA[2][128 * BKn];   // 8KB per buffer
  __shared__ unsigned short sB[2][64 * BKn];    // 4KB per buffer

  const int tid  = threadIdx.x;
  const int lane = tid & 31;
  const int wave = tid >> 5;
  const int hf   = lane >> 4;           // lane half (K split)
  const int l15  = lane & 15;
  const int rowBlk = blockIdx.y * 128;
  const int colBlk = blockIdx.x * 64;
  const int wm = (wave >> 1) * 32;
  const int wn = (wave & 1) * 32;
  const int NK = K / BKn;

  // Per-stage copies: A tile = 512 x 16B chunks (2/thread), B tile = 256 (1/thread)
  auto issue = [&](int ks, int buf) {
    const int kk = ks * BKn;
#pragma unroll
    for (int u = 0; u < 2; ++u) {
      int ch  = tid + u * 256;
      int r   = ch >> 2;
      int c16 = (ch & 3) * 8;
      async_copy_b128(A + (size_t)(rowBlk + r) * K + kk + c16,
                      &sA[buf][r * BKn + c16]);
    }
    {
      int r   = tid >> 2;
      int c16 = (tid & 3) * 8;
      async_copy_b128(Bw + (size_t)(colBlk + r) * K + kk + c16,
                      &sB[buf][r * BKn + c16]);
    }
  };

  v8f acc[2][2] = {};
  issue(0, 0);
  for (int ks = 0; ks < NK; ++ks) {
    const int buf = ks & 1;
    if (ks + 1 < NK) { issue(ks + 1, buf ^ 1); wait_async_3(); }
    else             { wait_async_0(); }
    __syncthreads();                     // stage ks fully resident in LDS

    v16bf a[2], b[2];
#pragma unroll
    for (int i = 0; i < 2; ++i) {
      // A fragment 16x32: lane=row, K chunks {hf*8..+7, 16+hf*8..+7}
      const unsigned short* pa = &sA[buf][(wm + i * 16 + l15) * BKn + hf * 8];
      a[i] = ldfrag(pa, pa + 16);
      // B fragment 32x16: lane=col, K chunks {hf*8..+7, 16+hf*8..+7}
      const unsigned short* pb = &sB[buf][(wn + i * 16 + l15) * BKn + hf * 8];
      b[i] = ldfrag(pb, pb + 16);
    }
#pragma unroll
    for (int i = 0; i < 2; ++i)
#pragma unroll
      for (int j = 0; j < 2; ++j)
        acc[i][j] = __builtin_amdgcn_wmma_f32_16x16x32_bf16(
            false, a[i], false, b[j], (short)0, acc[i][j], false, false);

    __syncthreads();                     // reads done before buf is overwritten
  }

  // Epilogue: C/D layout -> VGPR r holds rows r (lanes 0-15) / r+8 (lanes 16-31)
#pragma unroll
  for (int i = 0; i < 2; ++i)
#pragma unroll
    for (int j = 0; j < 2; ++j) {
      int col = colBlk + wn + j * 16 + l15;
      float bv = bias ? bias[col] : 0.f;
#pragma unroll
      for (int r = 0; r < 8; ++r) {
        int row = rowBlk + wm + i * 16 + hf * 8 + r;
        float v = acc[i][j][r] + bv;
        size_t idx = (size_t)row * N + col;
        if (mode == 0) {
          outBf[idx] = f2bf(v);
        } else if (mode == 1) {
          int b  = row >> 14;           // row / HWn
          int hw = row & (HWn - 1);
          float x2 = v * gamma[col] + resid[((size_t)b * Cn + col) * HWn + hw];
          outF[idx]  = x2;
          outBf[idx] = f2bf(x2);
        } else if (mode == 2) {
          float g = 0.5f * v * (1.0f + erff(v * 0.70710678118654752f));
          outBf[idx] = f2bf(g);
        } else {
          outF[idx] = v;
        }
      }
    }
}

// ---------------------------------------------------------------------------
// Axial attention, one workgroup per (b, head, fixed row/col).
//   axis 0: attend along W (fixed h);  axis 1: attend along H (fixed w)
//   qkv: bf16 [pos, 2304] with head layout [q96|k96|v96] per head block of 288
//   LDS: sV (V^T, 96x128 bf16) + region reused as {sQ,sK} then sP (P, 128x128)
// ---------------------------------------------------------------------------
__launch_bounds__(256)
__global__ void k_attn(const unsigned short* __restrict__ qkv,
                       float* __restrict__ att,
                       const float* __restrict__ qnw, const float* __restrict__ qnb,
                       const float* __restrict__ knw, const float* __restrict__ knb,
                       int axis, int accumulate) {
  __shared__ unsigned short sV[HEADn * Ln];       // 12288 (V transposed: [c][key])
  __shared__ unsigned short sX[2 * Ln * HEADn];   // 24576: sQ|sK, later sP(16384)
  unsigned short* sQ = sX;
  unsigned short* sK = sX + Ln * HEADn;
  unsigned short* sP = sX;                        // overlays sQ/sK after barrier

  const int b  = blockIdx.z;
  const int he = blockIdx.y;
  const int fx = blockIdx.x;

  size_t qBase, oBase; int strQ, strO;
  if (axis == 0) {  // along W, fixed h = fx
    qBase = ((size_t)b * HWn + (size_t)fx * Wn) * QKVCn + (size_t)he * 3 * HEADn;
    strQ  = QKVCn;
    oBase = ((size_t)b * HWn + (size_t)fx * Wn) * Cn + (size_t)he * HEADn;
    strO  = Cn;
  } else {          // along H, fixed w = fx
    qBase = ((size_t)b * HWn + fx) * QKVCn + (size_t)he * 3 * HEADn;
    strQ  = Wn * QKVCn;
    oBase = ((size_t)b * HWn + fx) * Cn + (size_t)he * HEADn;
    strO  = Wn * Cn;
  }

  const int tid = threadIdx.x;
  // --- stage Q,K with LayerNorm(96, eps=1e-5) as bf16 ---
  {
    int r   = tid & 127;
    int isK = tid >> 7;
    const unsigned short* src = qkv + qBase + (size_t)r * strQ + isK * HEADn;
    float s = 0.f, s2 = 0.f;
    for (int c = 0; c < HEADn; ++c) { float v = bf2f(src[c]); s += v; s2 += v * v; }
    float mu  = s * (1.0f / HEADn);
    float var = s2 * (1.0f / HEADn) - mu * mu;
    float inv = rsqrtf(var + 1e-5f);
    const float* wv = isK ? knw : qnw;
    const float* bb = isK ? knb : qnb;
    unsigned short* dst = (isK ? sK : sQ) + r * HEADn;
    for (int c = 0; c < HEADn; ++c)
      dst[c] = f2bf((bf2f(src[c]) - mu) * inv * wv[c] + bb[c]);
  }
  // --- stage V transposed: sV[c][key] ---
  if (tid < Ln) {
    int r = tid;
    const unsigned short* src = qkv + qBase + (size_t)r * strQ + 2 * HEADn;
    for (int c = 0; c < HEADn; ++c) sV[c * Ln + r] = src[c];
  }
  __syncthreads();

  const int lane = tid & 31;
  const int wave = tid >> 5;
  const int hf   = lane >> 4;
  const int l15  = lane & 15;
  const int m0   = wave * 16;           // this wave's 16 query rows

  // --- S = Q K^T : 1x8 tiles of 16x16, K(dim)=96 in 3 steps of 32 ---
  v8f acc[8] = {};
#pragma unroll
  for (int ks = 0; ks < 3; ++ks) {
    int kk = ks * 32;
    const unsigned short* pa = sQ + (m0 + l15) * HEADn + kk + hf * 8;
    v16bf a = ldfrag(pa, pa + 16);
#pragma unroll
    for (int t = 0; t < 8; ++t) {
      const unsigned short* pb = sK + (t * 16 + l15) * HEADn + kk + hf * 8;
      v16bf bfr = ldfrag(pb, pb + 16);
      acc[t] = __builtin_amdgcn_wmma_f32_16x16x32_bf16(
          false, a, false, bfr, (short)0, acc[t], false, false);
    }
  }

  // --- softmax along keys (row stats live in 16-lane halves) ---
  const float scale = 0.1020620726159658f;  // 1/sqrt(96)
  float invRow[8];
#pragma unroll
  for (int i = 0; i < 8; ++i) {
    float mx = -1e30f;
#pragma unroll
    for (int t = 0; t < 8; ++t) mx = fmaxf(mx, acc[t][i]);
#pragma unroll
    for (int d = 1; d < 16; d <<= 1) mx = fmaxf(mx, __shfl_xor(mx, d, 32));
    float sum = 0.f;
#pragma unroll
    for (int t = 0; t < 8; ++t) {
      float p = __expf((acc[t][i] - mx) * scale);
      acc[t][i] = p; sum += p;
    }
#pragma unroll
    for (int d = 1; d < 16; d <<= 1) sum += __shfl_xor(sum, d, 32);
    invRow[i] = 1.f / sum;
  }

  __syncthreads();  // everyone done reading sQ/sK -> safe to overlay sP

  // --- spill normalized P (bf16) to LDS in A-matrix-friendly row-major ---
#pragma unroll
  for (int t = 0; t < 8; ++t)
#pragma unroll
    for (int i = 0; i < 8; ++i)
      sP[(m0 + hf * 8 + i) * Ln + t * 16 + l15] = f2bf(acc[t][i] * invRow[i]);

  // --- O = P V : 1x6 tiles (96 channels), K=128 in 4 steps of 32 ---
  v8f o[6] = {};
#pragma unroll
  for (int ks = 0; ks < 4; ++ks) {
    int kk = ks * 32;
    const unsigned short* pa = sP + (m0 + l15) * Ln + kk + hf * 8;
    v16bf a = ldfrag(pa, pa + 16);
#pragma unroll
    for (int n = 0; n < 6; ++n) {
      const unsigned short* pb = sV + (n * 16 + l15) * Ln + kk + hf * 8;
      v16bf bfr = ldfrag(pb, pb + 16);
      o[n] = __builtin_amdgcn_wmma_f32_16x16x32_bf16(
          false, a, false, bfr, (short)0, o[n], false, false);
    }
  }

  // --- write 0.5 * contribution ---
#pragma unroll
  for (int n = 0; n < 6; ++n)
#pragma unroll
    for (int i = 0; i < 8; ++i) {
      int q = m0 + hf * 8 + i;
      size_t idx = oBase + (size_t)q * strO + n * 16 + l15;
      float v = 0.5f * o[n][i];
      if (accumulate) att[idx] += v; else att[idx] = v;
    }
}

// ---------------------------------------------------------------------------
// Final: out(bchw) = x2 + gamma_mlp * rms_in(m, mlp_norm_w)
// ---------------------------------------------------------------------------
__launch_bounds__(256)
__global__ void k_final(const float* __restrict__ x2,
                        const float* __restrict__ m,
                        const float* __restrict__ stds,
                        const float* __restrict__ mlpw,
                        const float* __restrict__ gmlp,
                        float* __restrict__ out) {
  long long i = (long long)blockIdx.x * 256 + threadIdx.x;
  if (i >= (long long)Bn * Cn * HWn) return;
  int hw = (int)(i % HWn);
  long long t = i / HWn;
  int c = (int)(t % Cn);
  int b = (int)(t / Cn);
  size_t pc = ((size_t)b * HWn + hw) * Cn + c;
  float mm = m[pc] / (stds[b * Cn + c] + 1e-8f) * mlpw[c];
  out[i] = x2[pc] + gmlp[c] * mm;
}

// ---------------------------------------------------------------------------
// Launch
// ---------------------------------------------------------------------------
extern "C" void kernel_launch(void* const* d_in, const int* in_sizes, int n_in,
                              void* d_out, int out_size, void* d_ws, size_t ws_size,
                              hipStream_t stream) {
  const float* x        = (const float*)d_in[0];
  const float* norm1_w  = (const float*)d_in[1];
  const float* qkv_w    = (const float*)d_in[2];
  const float* qkv_b    = (const float*)d_in[3];
  const float* qn_w     = (const float*)d_in[4];
  const float* qn_b     = (const float*)d_in[5];
  const float* kn_w     = (const float*)d_in[6];
  const float* kn_b     = (const float*)d_in[7];
  const float* norm2_w  = (const float*)d_in[8];
  const float* out_w    = (const float*)d_in[9];
  const float* out_b    = (const float*)d_in[10];
  const float* g_att    = (const float*)d_in[11];
  const float* fc1_w    = (const float*)d_in[12];
  const float* fc1_b    = (const float*)d_in[13];
  const float* fc2_w    = (const float*)d_in[14];
  const float* fc2_b    = (const float*)d_in[15];
  const float* mlp_w    = (const float*)d_in[16];
  const float* g_mlp    = (const float*)d_in[17];
  float* outp = (float*)d_out;

  char* ws = (char*)d_ws;
  size_t off = 0;
  auto take = [&](size_t bytes) -> char* {
    char* p = ws + off;
    off += (bytes + 255) & ~(size_t)255;
    return p;
  };
  float* std1 = (float*)take((size_t)Bn * Cn * 4);
  float* std2 = (float*)take((size_t)Bn * Cn * 4);
  float* std3 = (float*)take((size_t)Bn * Cn * 4);
  unsigned short* wqkv  = (unsigned short*)take((size_t)QKVCn * Cn * 2);
  unsigned short* woutw = (unsigned short*)take((size_t)Cn * Cn * 2);
  unsigned short* wfc1  = (unsigned short*)take((size_t)FFn * Cn * 2);
  unsigned short* wfc2  = (unsigned short*)take((size_t)Cn * FFn * 2);
  unsigned short* h1bf  = (unsigned short*)take((size_t)Pn * Cn * 2);    // also a_norm
  unsigned short* qkvbf = (unsigned short*)take((size_t)Pn * QKVCn * 2); // also m f32
  float*          attf  = (float*)take((size_t)Pn * Cn * 4);             // also x2 f32
  unsigned short* x2bf  = (unsigned short*)take((size_t)Pn * Cn * 2);
  unsigned short* m1bf  = (unsigned short*)take((size_t)Pn * FFn * 2);
  float* x2f  = attf;              // alias: att dead once a_norm exists
  float* mf32 = (float*)qkvbf;     // alias: qkv dead once attention is done

  // 1. weights -> bf16
  k_cvt_bf16<<<(QKVCn * Cn + 255) / 256, 256, 0, stream>>>(qkv_w, wqkv,  QKVCn * Cn);
  k_cvt_bf16<<<(Cn * Cn   + 255) / 256, 256, 0, stream>>>(out_w, woutw, Cn * Cn);
  k_cvt_bf16<<<(FFn * Cn  + 255) / 256, 256, 0, stream>>>(fc1_w, wfc1,  FFn * Cn);
  k_cvt_bf16<<<(Cn * FFn  + 255) / 256, 256, 0, stream>>>(fc2_w, wfc2,  Cn * FFn);

  // 2. norm1 -> h1 (bf16, [pos,C])
  k_std<<<Bn * Cn, 256, 0, stream>>>(x, std1, 0);
  long long nelem = (long long)Bn * Cn * HWn;
  int gE = (int)((nelem + 255) / 256);
  k_norm_bchw<<<gE, 256, 0, stream>>>(x, std1, norm1_w, h1bf);

  // 3. qkv GEMM (M=32768, N=2304, K=768) -> bf16
  k_gemm<<<dim3(QKVCn / 64, Pn / 128), 256, 0, stream>>>(
      h1bf, wqkv, qkv_b, nullptr, qkvbf, nullptr, nullptr, Pn, QKVCn, Cn, 0);

  // 4. axial attention: along W then along H (accumulate)
  k_attn<<<dim3(Hn, NHn, Bn), 256, 0, stream>>>(qkvbf, attf, qn_w, qn_b, kn_w, kn_b, 0, 0);
  k_attn<<<dim3(Wn, NHn, Bn), 256, 0, stream>>>(qkvbf, attf, qn_w, qn_b, kn_w, kn_b, 1, 1);

  // 5. norm2 -> a_norm bf16 (reuses h1 region)
  k_std<<<Bn * Cn, 256, 0, stream>>>(attf, std2, 1);
  int gPC = (int)(((long long)Pn * Cn + 255) / 256);
  k_norm_posC<<<gPC, 256, 0, stream>>>(attf, std2, norm2_w, h1bf);

  // 6. out proj GEMM + gamma_att*. + residual -> x2 (f32 reuses att region, bf16)
  k_gemm<<<dim3(Cn / 64, Pn / 128), 256, 0, stream>>>(
      h1bf, woutw, out_b, x2f, x2bf, g_att, x, Pn, Cn, Cn, 1);

  // 7. fc1 GEMM + exact GELU -> bf16
  k_gemm<<<dim3(FFn / 64, Pn / 128), 256, 0, stream>>>(
      x2bf, wfc1, fc1_b, nullptr, m1bf, nullptr, nullptr, Pn, FFn, Cn, 2);

  // 8. fc2 GEMM -> f32 (reuses qkv region)
  k_gemm<<<dim3(Cn / 64, Pn / 128), 256, 0, stream>>>(
      m1bf, wfc2, fc2_b, mf32, nullptr, nullptr, nullptr, Pn, Cn, FFn, 3);

  // 9. mlp norm + final combine -> d_out (bchw)
  k_std<<<Bn * Cn, 256, 0, stream>>>(mf32, std3, 1);
  k_final<<<gE, 256, 0, stream>>>(x2f, mf32, std3, mlp_w, g_mlp, outp);
}
